// MemorizingTransformer_47467978555703
// MI455X (gfx1250) — compile-verified, matrix operand order checked by reading, not run on hardware
//
#include <hip/hip_runtime.h>
#include <math.h>

// ---------------- model constants ----------------
constexpr int B_ = 2, N_ = 2048, DIM_ = 1024, HEADS_ = 8, DH_ = 64;
constexpr int DEPTH_ = 4, VOCAB_ = 32000, FFD_ = 4096, INNER_ = 512;
constexpr int ROWS_ = B_ * N_;            // 4096 token rows

typedef __attribute__((ext_vector_type(16))) __bf16 bf16x16;
typedef __attribute__((ext_vector_type(8)))  float  f32x8;

struct alignas(16) U4 { unsigned x, y, z, w; };
struct alignas(8)  U2 { unsigned x, y; };
union FragBF { bf16x16 v; __bf16 e[16]; U4 q[2]; };
union FragF  { f32x8   v; float  e[8]; };
union Pack8  { __bf16 e[8]; U4 q; };
union Pack4  { __bf16 e[4]; U2 q; };

#define DEVI static __device__ __forceinline__

DEVI __bf16 f2bf(float f) { return (__bf16)f; }   // hardware cvt (RNE)

DEVI f32x8 wmma_bf16(bf16x16 a, bf16x16 b, f32x8 c) {
  return __builtin_amdgcn_wmma_f32_16x16x32_bf16(false, a, false, b,
                                                 (short)0, c, false, false);
}

// ---------------- embedding gather ----------------
__global__ __launch_bounds__(256)
void embed_kernel(const int* __restrict__ tok, const float* __restrict__ emb,
                  float* __restrict__ x) {
  const int row = blockIdx.x;
  const int t = tok[row];
  const float4* src = (const float4*)(emb + (size_t)t * DIM_);
  float4* dst = (float4*)(x + (size_t)row * DIM_);
  dst[threadIdx.x] = src[threadIdx.x];          // 256 thr * float4 = 1024
}

// ---------------- layernorm: f32 -> bf16 ----------------
__global__ __launch_bounds__(256)
void layernorm_kernel(const float* __restrict__ x, const float* __restrict__ g,
                      const float* __restrict__ bta, __bf16* __restrict__ out) {
  const int row = blockIdx.x;
  const float* xr = x + (size_t)row * DIM_;
  float4 v4 = ((const float4*)xr)[threadIdx.x];
  float s  = v4.x + v4.y + v4.z + v4.w;
  float s2 = v4.x*v4.x + v4.y*v4.y + v4.z*v4.z + v4.w*v4.w;
  #pragma unroll
  for (int m = 16; m >= 1; m >>= 1) { s += __shfl_xor(s, m); s2 += __shfl_xor(s2, m); }
  __shared__ float ws[8], ws2[8];
  const int wave = threadIdx.x >> 5, lane = threadIdx.x & 31;
  if (lane == 0) { ws[wave] = s; ws2[wave] = s2; }
  __syncthreads();
  s = 0.f; s2 = 0.f;
  #pragma unroll
  for (int w = 0; w < 8; ++w) { s += ws[w]; s2 += ws2[w]; }
  const float mu  = s * (1.f / DIM_);
  const float var = s2 * (1.f / DIM_) - mu * mu;
  const float inv = rsqrtf(var + 1e-5f);
  const int d = threadIdx.x * 4;
  const float vv[4] = { v4.x, v4.y, v4.z, v4.w };
  Pack4 o;
  #pragma unroll
  for (int j = 0; j < 4; ++j)
    o.e[j] = f2bf((vv[j] - mu) * inv * g[d + j] + bta[d + j]);
  *(U2*)(out + (size_t)row * DIM_ + d) = o.q;
}

// ---------------- weight transpose + cast: W(KxN) f32 -> Wt(NxK) bf16 ------
__global__ __launch_bounds__(256)
void transpose_cast(const float* __restrict__ W, __bf16* __restrict__ Wt,
                    int K, int N) {
  __shared__ float tile[32][33];
  const int kb = blockIdx.y * 32, nb = blockIdx.x * 32;
  const int tx = threadIdx.x & 31, ty = threadIdx.x >> 5;   // ty 0..7
  #pragma unroll
  for (int r = ty; r < 32; r += 8)
    tile[r][tx] = W[(size_t)(kb + r) * N + nb + tx];
  __syncthreads();
  #pragma unroll
  for (int r = ty; r < 32; r += 8)
    Wt[(size_t)(nb + r) * K + kb + tx] = f2bf(tile[tx][r]);
}

// ---------------- V transpose: kvb(B*N x 128)[:,64:] -> vt[b][dh][key] -----
__global__ __launch_bounds__(256)
void vtrans_kernel(const __bf16* __restrict__ kvb, __bf16* __restrict__ vt) {
  __shared__ __bf16 tile[32][33];
  const int jb = blockIdx.x * 32;        // key base
  const int db = blockIdx.y * 32;        // dh base
  const int bb = blockIdx.z;
  const int tx = threadIdx.x & 31, ty = threadIdx.x >> 5;
  #pragma unroll
  for (int r = ty; r < 32; r += 8)
    tile[r][tx] = kvb[(size_t)(bb*N_ + jb + r) * (2*DH_) + DH_ + db + tx];
  __syncthreads();
  #pragma unroll
  for (int r = ty; r < 32; r += 8)
    vt[(size_t)(bb*DH_ + db + r) * N_ + jb + tx] = tile[tx][r];
}

// ---------------- rel-position bias table: tab[head][dist] -----------------
__global__ __launch_bounds__(256)
void relbias_kernel(const float* __restrict__ rel, float* __restrict__ tab) {
  const int d = blockIdx.x * 256 + threadIdx.x;   // distance 0..N-1
  const int h = blockIdx.y;
  int bucket;
  if (d < 16) bucket = d;
  else {
    bucket = 16 + (int)(logf((float)d * 0.0625f) * 7.69450073f);  // 16/ln8
    bucket = bucket > 31 ? 31 : bucket;
  }
  tab[(size_t)h * N_ + d] = rel[bucket * HEADS_ + h] * 8.0f;      // * sqrt(DH)
}

// ---------------- bf16 WMMA GEMM --------------------------------------------
// C(MxN) = A(MxK, bf16 row-major) * Bt(NxK, bf16 row-major = B transposed)
// Workgroup tile 256(M) x 128(N); 8 waves as 4x2, each wave 64x64
// (16 WMMAs per 8 fragment loads per K-step: 1.0 WMMA per b128 load)
// EPI bit0: +bias[n]  bit1: +resid[m,n]  bit2: exact GELU  bit3: store bf16
// EPI bit4: pre-scale by DH^-0.5 (for Q projection)
template<int EPI>
__global__ __launch_bounds__(256)
void gemm_bf16(const __bf16* __restrict__ A, const __bf16* __restrict__ Bt,
               const float* __restrict__ bias, const float* __restrict__ resid,
               float* __restrict__ Cf, __bf16* __restrict__ Cb,
               int M, int N, int K) {
  const int wave = threadIdx.x >> 5;
  const int lane = threadIdx.x & 31;
  const int lr = lane & 15, half = lane >> 4;
  const int m0 = blockIdx.y * 256 + (wave & 3) * 64;   // 4 M-frags per wave
  const int n0 = blockIdx.x * 128 + (wave >> 2) * 64;  // 4 N-frags per wave
  f32x8 acc[4][4] = {};
  for (int k0 = 0; k0 < K; k0 += 32) {
    FragBF a[4], b[4];
    #pragma unroll
    for (int i = 0; i < 4; ++i) {
      const __bf16* p = A + (size_t)(m0 + i*16 + lr) * K + k0 + half*8;
      a[i].q[0] = *(const U4*)p;
      a[i].q[1] = *(const U4*)(p + 16);      // K offset +16
    }
    #pragma unroll
    for (int j = 0; j < 4; ++j) {
      const __bf16* p = Bt + (size_t)(n0 + j*16 + lr) * K + k0 + half*8;
      b[j].q[0] = *(const U4*)p;
      b[j].q[1] = *(const U4*)(p + 16);
    }
    #pragma unroll
    for (int i = 0; i < 4; ++i)
      #pragma unroll
      for (int j = 0; j < 4; ++j)
        acc[i][j] = wmma_bf16(a[i].v, b[j].v, acc[i][j]);
  }
  #pragma unroll
  for (int i = 0; i < 4; ++i)
    #pragma unroll
    for (int j = 0; j < 4; ++j) {
      FragF c; c.v = acc[i][j];
      #pragma unroll
      for (int r = 0; r < 8; ++r) {
        const int m = m0 + i*16 + half*8 + r;
        const int n = n0 + j*16 + lr;
        float v = c.e[r];
        if constexpr ((EPI & 16) != 0) v *= 0.125f;                 // DH^-0.5
        if constexpr ((EPI & 1)  != 0) v += bias[n];
        if constexpr ((EPI & 4)  != 0) v = 0.5f * v * (1.f + erff(v * 0.70710678f));
        if constexpr ((EPI & 2)  != 0) v += resid[(size_t)m * N + n];
        if constexpr ((EPI & 8)  != 0) Cb[(size_t)m * N + n] = f2bf(v);
        else                           Cf[(size_t)m * N + n] = v;
      }
    }
}

// ---------------- flash attention (MQA, rel-bucket bias, causal) ------------
// one wave handles (batch, head, 32-query tile); S^T = K*Q^T, O^T = V^T*P^T
// q, kv, vt already bf16 (q pre-scaled by DH^-0.5 in its GEMM epilogue)
__global__ __launch_bounds__(32)
void attn_kernel(const __bf16* __restrict__ q, const __bf16* __restrict__ kvb,
                 const __bf16* __restrict__ vt, const float* __restrict__ rtab,
                 __bf16* __restrict__ obf) {
  const int lane = threadIdx.x;
  const int lr = lane & 15, half = lane >> 4;
  const int qbase = blockIdx.x * 32;
  const int head  = blockIdx.y;
  const int bb    = blockIdx.z;
  const float* tab = rtab + (size_t)head * N_;     // bias by distance i-j

  // Q^T B-fragments: [query-frag nq][dh-frag kf] -- straight b128 loads
  FragBF qb[2][2];
  #pragma unroll
  for (int nq = 0; nq < 2; ++nq)
    #pragma unroll
    for (int kf = 0; kf < 2; ++kf) {
      const __bf16* p = q + (size_t)(bb*N_ + qbase + nq*16 + lr) * INNER_
                          + head*DH_ + kf*32 + half*8;
      qb[nq][kf].q[0] = *(const U4*)p;
      qb[nq][kf].q[1] = *(const U4*)(p + 16);
    }

  f32x8 oacc[4][2] = {};                 // O^T: [dh-frag md][query-frag nq]
  float mrun[2] = { -3.0e38f, -3.0e38f };
  float lrun[2] = { 0.f, 0.f };

  for (int j0 = 0; j0 <= qbase; j0 += 32) {
    // ---- S^T tiles (M=key, N=query) via WMMA(A=K, B=Q^T) ----
    f32x8 st[2][2] = {};
    #pragma unroll
    for (int mk = 0; mk < 2; ++mk) {
      FragBF ka[2];
      const __bf16* p = kvb + (size_t)(bb*N_ + j0 + mk*16 + lr) * (2*DH_) + half*8;
      #pragma unroll
      for (int kf = 0; kf < 2; ++kf) {
        ka[kf].q[0] = *(const U4*)(p + kf*32);
        ka[kf].q[1] = *(const U4*)(p + kf*32 + 16);
      }
      #pragma unroll
      for (int nq = 0; nq < 2; ++nq)
        #pragma unroll
        for (int kf = 0; kf < 2; ++kf)
          st[mk][nq] = wmma_bf16(ka[kf].v, qb[nq][kf].v, st[mk][nq]);
    }

    // ---- bias (table lookup) + causal mask + chunk max ----
    float sv[2][2][8];
    float mc[2] = { -3.0e38f, -3.0e38f };
    #pragma unroll
    for (int mk = 0; mk < 2; ++mk) {
      FragF c0, c1; c0.v = st[mk][0]; c1.v = st[mk][1];
      #pragma unroll
      for (int r = 0; r < 8; ++r) {
        const int j = j0 + mk*16 + half*8 + r;       // key index
        #pragma unroll
        for (int nq = 0; nq < 2; ++nq) {
          const int i = qbase + nq*16 + lr;          // query index
          const int n = i - j;                       // distance (>=0 when valid)
          const float bv = tab[n < 0 ? 0 : n];
          const float raw = (nq == 0 ? c0.e[r] : c1.e[r]);
          const float s = (n >= 0) ? raw + bv : -3.0e38f;
          sv[mk][nq][r] = s;
          mc[nq] = fmaxf(mc[nq], s);
        }
      }
    }
    #pragma unroll
    for (int nq = 0; nq < 2; ++nq) mc[nq] = fmaxf(mc[nq], __shfl_xor(mc[nq], 16));

    // ---- online softmax update ----
    float corr[2], lc[2];
    #pragma unroll
    for (int nq = 0; nq < 2; ++nq) {
      const float mnew = fmaxf(mrun[nq], mc[nq]);
      corr[nq] = __expf(mrun[nq] - mnew);
      mrun[nq] = mnew;
      lc[nq] = 0.f;
    }
    FragBF pb[2];        // P^T B-fragments: key r->e, half->half (no shuffles)
    #pragma unroll
    for (int nq = 0; nq < 2; ++nq)
      #pragma unroll
      for (int r = 0; r < 8; ++r) {
        const float p0 = __expf(sv[0][nq][r] - mrun[nq]);
        const float p1 = __expf(sv[1][nq][r] - mrun[nq]);
        pb[nq].e[r]     = f2bf(p0);
        pb[nq].e[r + 8] = f2bf(p1);
        lc[nq] += p0 + p1;
      }
    #pragma unroll
    for (int nq = 0; nq < 2; ++nq) {
      lc[nq] += __shfl_xor(lc[nq], 16);
      lrun[nq] = lrun[nq] * corr[nq] + lc[nq];
    }
    #pragma unroll
    for (int md = 0; md < 4; ++md)
      #pragma unroll
      for (int nq = 0; nq < 2; ++nq) {
        FragF c; c.v = oacc[md][nq];
        #pragma unroll
        for (int r = 0; r < 8; ++r) c.e[r] *= corr[nq];
        oacc[md][nq] = c.v;
      }

    // ---- O^T += V^T * P^T  (V^T pre-transposed: b128 loads) ----
    #pragma unroll
    for (int md = 0; md < 4; ++md) {
      FragBF va;
      const __bf16* pv = vt + (size_t)(bb*DH_ + md*16 + lr) * N_ + j0 + half*8;
      va.q[0] = *(const U4*)pv;
      va.q[1] = *(const U4*)(pv + 16);
      #pragma unroll
      for (int nq = 0; nq < 2; ++nq)
        oacc[md][nq] = wmma_bf16(va.v, pb[nq].v, oacc[md][nq]);
    }
  }

  // ---- normalize & store O as bf16 (B,N,INNER): one b128 store per tile ----
  #pragma unroll
  for (int md = 0; md < 4; ++md)
    #pragma unroll
    for (int nq = 0; nq < 2; ++nq) {
      FragF c; c.v = oacc[md][nq];
      const float inv = 1.0f / lrun[nq];
      const int query = qbase + nq*16 + lr;
      Pack8 o8;
      #pragma unroll
      for (int r = 0; r < 8; ++r) o8.e[r] = f2bf(c.e[r] * inv);
      *(U4*)(obf + (size_t)(bb*N_ + query) * INNER_ + head*DH_ + md*16 + half*8) = o8.q;
    }
}

// ---------------- orchestration ----------------
extern "C" void kernel_launch(void* const* d_in, const int* in_sizes, int n_in,
                              void* d_out, int out_size, void* d_ws, size_t ws_size,
                              hipStream_t stream) {
  const int*   tok     = (const int*)  d_in[0];
  const float* tok_emb = (const float*)d_in[1];
  const float* ln1_g   = (const float*)d_in[2];
  const float* ln1_b   = (const float*)d_in[3];
  const float* Wq      = (const float*)d_in[4];
  const float* Wkv     = (const float*)d_in[5];
  const float* Wo      = (const float*)d_in[6];
  const float* bo      = (const float*)d_in[7];
  const float* rel_emb = (const float*)d_in[8];
  const float* ln2_g   = (const float*)d_in[9];
  const float* ln2_b   = (const float*)d_in[10];
  const float* W1      = (const float*)d_in[11];
  const float* b1      = (const float*)d_in[12];
  const float* W2      = (const float*)d_in[13];
  const float* b2      = (const float*)d_in[14];
  const float* lnf_g   = (const float*)d_in[15];
  const float* lnf_b   = (const float*)d_in[16];
  const float* Wl      = (const float*)d_in[17];
  const float* bl      = (const float*)d_in[18];
  float* out = (float*)d_out;

  // workspace layout (bytes)
  char* ws = (char*)d_ws;
  float*  x    = (float*) (ws + 0);                                  // 16 MB
  __bf16* hbf  = (__bf16*)(ws + (size_t)16777216);                   //  8 MB
  __bf16* qb   = (__bf16*)(ws + (size_t)25165824);                   //  4 MB
  __bf16* kvb  = (__bf16*)(ws + (size_t)29360128);                   //  1 MB
  __bf16* vt   = (__bf16*)(ws + (size_t)30408704);                   // 0.5 MB
  float*  rtab = (float*) (ws + (size_t)30932992);                   // 64 KB
  __bf16* obf  = (__bf16*)(ws + (size_t)30998528);                   //  4 MB
  __bf16* ffbf = (__bf16*)(ws + (size_t)35192832);                   // 32 MB
  __bf16* wt   = (__bf16*)(ws + (size_t)68747264);                   // 64 MB

  embed_kernel<<<ROWS_, 256, 0, stream>>>(tok, tok_emb, x);

  for (int l = 0; l < DEPTH_; ++l) {
    // LN1 -> h (bf16)
    layernorm_kernel<<<ROWS_, 256, 0, stream>>>(x, ln1_g + l*DIM_, ln1_b + l*DIM_, hbf);

    // q = (h @ Wq[l]) * DH^-0.5, stored bf16   (4096x1024 @ 1024x512)
    transpose_cast<<<dim3(INNER_/32, DIM_/32), 256, 0, stream>>>(Wq + (size_t)l*DIM_*INNER_, wt, DIM_, INNER_);
    gemm_bf16<24><<<dim3(INNER_/128, ROWS_/256), 256, 0, stream>>>(hbf, wt, nullptr, nullptr, nullptr, qb, ROWS_, INNER_, DIM_);

    // kv = h @ Wkv[l], stored bf16             (4096x1024 @ 1024x128)
    transpose_cast<<<dim3((2*DH_)/32, DIM_/32), 256, 0, stream>>>(Wkv + (size_t)l*DIM_*2*DH_, wt, DIM_, 2*DH_);
    gemm_bf16<8><<<dim3((2*DH_)/128, ROWS_/256), 256, 0, stream>>>(hbf, wt, nullptr, nullptr, nullptr, kvb, ROWS_, 2*DH_, DIM_);

    // V transpose + rel-bias table + attention -> o (bf16)
    vtrans_kernel<<<dim3(N_/32, DH_/32, B_), 256, 0, stream>>>(kvb, vt);
    relbias_kernel<<<dim3(N_/256, HEADS_), 256, 0, stream>>>(rel_emb + (size_t)l*32*HEADS_, rtab);
    attn_kernel<<<dim3(N_/32, HEADS_, B_), 32, 0, stream>>>(qb, kvb, vt, rtab, obf);

    // x = o @ Wo[l] + bo + x
    transpose_cast<<<dim3(DIM_/32, INNER_/32), 256, 0, stream>>>(Wo + (size_t)l*INNER_*DIM_, wt, INNER_, DIM_);
    gemm_bf16<3><<<dim3(DIM_/128, ROWS_/256), 256, 0, stream>>>(obf, wt, bo + l*DIM_, x, x, nullptr, ROWS_, DIM_, INNER_);

    // LN2 -> h2 (bf16)
    layernorm_kernel<<<ROWS_, 256, 0, stream>>>(x, ln2_g + l*DIM_, ln2_b + l*DIM_, hbf);

    // ff = gelu(h2 @ W1 + b1) (bf16)
    transpose_cast<<<dim3(FFD_/32, DIM_/32), 256, 0, stream>>>(W1 + (size_t)l*DIM_*FFD_, wt, DIM_, FFD_);
    gemm_bf16<13><<<dim3(FFD_/128, ROWS_/256), 256, 0, stream>>>(hbf, wt, b1 + l*FFD_, nullptr, nullptr, ffbf, ROWS_, FFD_, DIM_);

    // x = ff @ W2 + b2 + x
    transpose_cast<<<dim3(DIM_/32, FFD_/32), 256, 0, stream>>>(W2 + (size_t)l*FFD_*DIM_, wt, FFD_, DIM_);
    gemm_bf16<3><<<dim3(DIM_/128, ROWS_/256), 256, 0, stream>>>(ffbf, wt, b2 + l*DIM_, x, x, nullptr, ROWS_, DIM_, FFD_);
  }

  // final LN -> bf16, then logits = xf @ Wl + bl  (4096x1024 @ 1024x32000)
  layernorm_kernel<<<ROWS_, 256, 0, stream>>>(x, lnf_g, lnf_b, hbf);
  transpose_cast<<<dim3(VOCAB_/32, DIM_/32), 256, 0, stream>>>(Wl, wt, DIM_, VOCAB_);
  gemm_bf16<1><<<dim3(VOCAB_/128, ROWS_/256), 256, 0, stream>>>(hbf, wt, bl, nullptr, out, nullptr, ROWS_, VOCAB_, DIM_);

  (void)in_sizes; (void)n_in; (void)out_size; (void)ws_size;
}
